// Network_6150393168677
// MI455X (gfx1250) — matrix operand order repaired and checked by reading, not compiled, not run on hardware
//
#include <hip/hip_runtime.h>
#include <stdint.h>

// ---------------------------------------------------------------------------
// Problem constants (from the reference)
// ---------------------------------------------------------------------------
#define B_  16
#define N_  1024
#define C_  32
#define K_  9
#define MROWS ((long long)B_ * N_)   // 16384

// ---------------------------------------------------------------------------
// WMMA types (CDNA5 / gfx1250, wave32)
// ---------------------------------------------------------------------------
typedef __attribute__((ext_vector_type(16))) __bf16 bf16x16;
typedef __attribute__((ext_vector_type(8)))  float  floatx8;

__device__ __forceinline__ unsigned int f2bf(float f) {
  // round-to-nearest-even fp32 -> bf16 (returned in low 16 bits)
  unsigned int u = __float_as_uint(f);
  unsigned int r = u + 0x7FFFu + ((u >> 16) & 1u);
  return r >> 16;
}

// ---------------------------------------------------------------------------
// Fragment-linear bf16 packing.
//
// A-pack:  Ap[(row*KB + kb)*16 + p] = bf16pair(A[row, kb*32+2p], A[row, kb*32+2p+1])
//   Lane (r,hi) of a 16x16 tile needs K pairs p = hi*4+0..3 and 8+hi*4+0..3
//   -> exactly two b128 loads per K-block.
// W-pack:  Wp[(kb*Np + n)*16 + p] = bf16pair(W[kb*32+2p, n], W[kb*32+2p+1, n])
//   Lane needs pairs p = hi*4+0..3 and 8+hi*4+0..3 -> two b128 loads per K-block.
// Both support zero-padding K (Ksrc<Kp) and N (Nsrc<Np).
// ---------------------------------------------------------------------------
__global__ void pack_a(const float* __restrict__ A, int Ksrc, int Kp,
                       long long M, unsigned int* __restrict__ Ap)
{
  const int KB = Kp >> 5;
  long long i = (long long)blockIdx.x * blockDim.x + threadIdx.x;
  long long tot = M * KB * 16;
  if (i >= tot) return;
  int p = (int)(i & 15);
  long long t = i >> 4;
  int kb = (int)(t % KB);
  long long row = t / KB;
  int k0 = kb * 32 + 2 * p;
  float lo, hv;
  if (Ksrc == Kp) {                    // uniform branch: unguarded loads
    lo = A[row * Ksrc + k0];
    hv = A[row * Ksrc + k0 + 1];
  } else {
    lo = (k0     < Ksrc) ? A[row * Ksrc + k0]     : 0.f;
    hv = (k0 + 1 < Ksrc) ? A[row * Ksrc + k0 + 1] : 0.f;
  }
  Ap[i] = f2bf(lo) | (f2bf(hv) << 16);
}

__global__ void pack_w(const float* __restrict__ W, int Ksrc, int Kp,
                       int Nsrc, int Np, unsigned int* __restrict__ Wp)
{
  const int KB = Kp >> 5;
  long long i = (long long)blockIdx.x * blockDim.x + threadIdx.x;
  long long tot = (long long)KB * Np * 16;
  if (i >= tot) return;
  int p = (int)(i & 15);
  long long t = i >> 4;
  int n = (int)(t % Np);
  int kb = (int)(t / Np);
  int k0 = kb * 32 + 2 * p;
  float lo, hv;
  if (Ksrc == Kp && Nsrc == Np) {      // uniform branch: unguarded loads
    lo = W[(long long)k0 * Nsrc + n];
    hv = W[(long long)(k0 + 1) * Nsrc + n];
  } else {
    lo = (k0     < Ksrc && n < Nsrc) ? W[(long long)k0 * Nsrc + n]       : 0.f;
    hv = (k0 + 1 < Ksrc && n < Nsrc) ? W[(long long)(k0 + 1) * Nsrc + n] : 0.f;
  }
  Wp[i] = f2bf(lo) | (f2bf(hv) << 16);
}

// ---------------------------------------------------------------------------
// Packed-bf16 WMMA GEMM:  Y[M,N] (+)= A[M,K] @ W[K,N]
// Requires M%16==0 (or wave-tiles beyond M unused), N%16==0, K%32==0.
// One 16x16 tile per wave, 4 waves per block along M. No per-lane guards:
// inner loop = 2x b128 (A) + 2x b128 (B) + v_wmma.
// ---------------------------------------------------------------------------
__global__ void gemm_wmma_packed(const unsigned int* __restrict__ Ap,
                                 const unsigned int* __restrict__ Wp,
                                 float* __restrict__ Y,
                                 int M, int K, int N, int accumulate)
{
  const int lane  = threadIdx.x;        // 0..31
  const int r     = lane & 15;
  const int hi    = lane >> 4;
  const int tileN = blockIdx.x;
  const int tileM = blockIdx.y * blockDim.y + threadIdx.y;
  if (tileM * 16 >= M) return;          // wave-uniform (M=16 head case)
  const int row = tileM * 16 + r;
  const int col = tileN * 16 + r;
  const int KB  = K >> 5;

  const uint4* __restrict__ AP4 = (const uint4*)Ap;
  const uint4* __restrict__ WP4 = (const uint4*)Wp;

  floatx8 acc = {0.f, 0.f, 0.f, 0.f, 0.f, 0.f, 0.f, 0.f};
  union { bf16x16 v; uint4 q[2]; } fa, fb;

  long long abase = (long long)row * KB * 4;            // uint4 units
  long long bbase = (long long)col * 4 + (long long)hi;
  for (int kb = 0; kb < KB; ++kb) {
    if (kb + 1 < KB) __builtin_prefetch(&AP4[abase + 4 + hi]); // global_prefetch_b8
    fa.q[0] = AP4[abase + hi];
    fa.q[1] = AP4[abase + 2 + hi];
    fb.q[0] = WP4[bbase];
    fb.q[1] = WP4[bbase + 2];
    acc = __builtin_amdgcn_wmma_f32_16x16x32_bf16(false, fa.v, false, fb.v,
                                                  (short)0, acc, false, false);
    abase += 4;
    bbase += (long long)N * 4;
  }

#pragma unroll
  for (int v = 0; v < 8; ++v) {
    int m = tileM * 16 + v + 8 * hi;
    float val = acc[v];
    if (accumulate) val += Y[(long long)m * N + col];
    Y[(long long)m * N + col] = val;
  }
}

// ---------------------------------------------------------------------------
// BatchNorm over all-but-last axis: per-channel mean / population variance
// ---------------------------------------------------------------------------
__global__ void bn_stats(const float* __restrict__ y, long long M, int C,
                         float* __restrict__ mv)
{
  int c = blockIdx.x;
  float s = 0.f, s2 = 0.f;
  for (long long i = threadIdx.x; i < M; i += blockDim.x) {
    float v = y[i * C + c];
    s += v; s2 += v * v;
  }
  __shared__ float sh1[256], sh2[256];
  sh1[threadIdx.x] = s; sh2[threadIdx.x] = s2;
  __syncthreads();
  for (int o = 128; o > 0; o >>= 1) {
    if ((int)threadIdx.x < o) {
      sh1[threadIdx.x] += sh1[threadIdx.x + o];
      sh2[threadIdx.x] += sh2[threadIdx.x + o];
    }
    __syncthreads();
  }
  if (threadIdx.x == 0) {
    float m = sh1[0] / (float)M;
    mv[2 * c]     = m;
    mv[2 * c + 1] = sh2[0] / (float)M - m * m;
  }
}

// act: 0=none, 1=relu, 2=leaky(0.2)
__global__ void bn_apply(float* __restrict__ y, long long M, int C,
                         const float* __restrict__ mv,
                         const float* __restrict__ g,
                         const float* __restrict__ b, int act)
{
  long long i = (long long)blockIdx.x * blockDim.x + threadIdx.x;
  if (i >= M * (long long)C) return;
  int c = (int)(i % C);
  float val = (y[i] - mv[2 * c]) * rsqrtf(mv[2 * c + 1] + 1e-5f) * g[c] + b[c];
  if (act == 1) val = fmaxf(val, 0.f);
  else if (act == 2) val = val > 0.f ? val : 0.2f * val;
  y[i] = val;
}

// ---------------------------------------------------------------------------
// kNN: per (b,n) block computes squared distances to all N points, then does
// kd iterative argmin selections (exact lax.top_k tie-breaking by low index),
// finally writes every dil-th selected index (dilation).
// ---------------------------------------------------------------------------
__global__ void rowsq_kernel(const float* __restrict__ s, int Cf,
                             float* __restrict__ sq)
{
  long long row = (long long)blockIdx.x * blockDim.x + threadIdx.x;
  if (row >= MROWS) return;
  const float* p = s + row * Cf;
  float a = 0.f;
  for (int c = 0; c < Cf; ++c) a += p[c] * p[c];
  sq[row] = a;
}

__global__ void knn_kernel(const float* __restrict__ s,
                           const float* __restrict__ sq,
                           int Cf, int kd, int dil, int* __restrict__ idx_out)
{
  const int n = blockIdx.x, b = blockIdx.y;
  const int tid = threadIdx.x;                 // blockDim.x == 128
  __shared__ float pnt[96];
  __shared__ float dist[N_];
  __shared__ float rv[128];
  __shared__ int   ri[128];
  __shared__ int   sel[32];

  const float* srow = s + ((long long)b * N_ + n) * Cf;
  for (int c = tid; c < Cf; c += blockDim.x) pnt[c] = srow[c];
  __syncthreads();

  const int Cf4 = Cf >> 2;   // Cf == 96, multiple of 4
  float sqn = sq[b * N_ + n];
  for (int m = tid; m < N_; m += blockDim.x) {
    const float4* sm4 = (const float4*)(s + ((long long)b * N_ + m) * Cf);
    float dot = 0.f;
    for (int c4 = 0; c4 < Cf4; ++c4) {
      float4 v = sm4[c4];
      dot += pnt[4 * c4] * v.x + pnt[4 * c4 + 1] * v.y +
             pnt[4 * c4 + 2] * v.z + pnt[4 * c4 + 3] * v.w;
    }
    dist[m] = sqn - 2.f * dot + sq[b * N_ + m];
  }
  __syncthreads();

  for (int t = 0; t < kd; ++t) {
    float best = 3.4e38f; int bi = N_;
    for (int m = tid; m < N_; m += blockDim.x) {
      float v = dist[m];
      if (v < best) { best = v; bi = m; }   // strided scan keeps lowest m on tie
    }
    rv[tid] = best; ri[tid] = bi;
    __syncthreads();
    for (int o = 64; o > 0; o >>= 1) {
      if (tid < o) {
        if (rv[tid + o] < rv[tid] ||
            (rv[tid + o] == rv[tid] && ri[tid + o] < ri[tid])) {
          rv[tid] = rv[tid + o]; ri[tid] = ri[tid + o];
        }
      }
      __syncthreads();
    }
    if (tid == 0) { sel[t] = ri[0]; dist[ri[0]] = 3.4e38f; }
    __syncthreads();
  }
  if (tid < K_) idx_out[((long long)b * N_ + n) * K_ + tid] = sel[tid * dil];
}

// ---------------------------------------------------------------------------
// Neighbor aggregation: nbsum = sum_k x[idx], maxdiff = max_k (x[idx]-x)
// block = (32 channels, 8 rows)
// ---------------------------------------------------------------------------
__global__ void nb_agg(const float* __restrict__ x, const int* __restrict__ idx,
                       float* __restrict__ nbsum, float* __restrict__ maxdiff)
{
  int c = threadIdx.x;
  long long row = (long long)blockIdx.x * blockDim.y + threadIdx.y;
  if (row >= MROWS) return;
  int b = (int)(row / N_);
  const int* ip = idx + row * K_;
  float xv = x[row * C_ + c];
  float s = 0.f, mx = -3.4e38f;
  for (int k = 0; k < K_; ++k) {
    float nv = x[((long long)b * N_ + ip[k]) * C_ + c];
    s += nv;
    mx = fmaxf(mx, nv - xv);
  }
  nbsum[row * C_ + c]   = s;
  maxdiff[row * C_ + c] = mx;
}

// EdgeConv pre-BN features, via gather identity:
// e[row,k,c] = (x@Wtop)[row,c] + (x@Wbot)[idx,c] - (x@Wbot)[row,c]
__global__ void edge_build(const float* __restrict__ tet,
                           const float* __restrict__ teb,
                           const int* __restrict__ idx, float* __restrict__ e)
{
  long long i = (long long)blockIdx.x * blockDim.x + threadIdx.x;
  long long tot = MROWS * K_ * C_;
  if (i >= tot) return;
  int c = (int)(i & (C_ - 1));
  long long t = i >> 5;
  int k = (int)(t % K_);
  long long row = t / K_;
  int b = (int)(row / N_);
  int nk = idx[row * K_ + k];
  e[i] = tet[row * C_ + c] + teb[((long long)b * N_ + nk) * C_ + c]
       - teb[row * C_ + c];
}

__global__ void edge_bn_max(const float* __restrict__ e,
                            const float* __restrict__ mv,
                            const float* __restrict__ g,
                            const float* __restrict__ bb,
                            float* __restrict__ out)
{
  int c = threadIdx.x;
  long long row = (long long)blockIdx.x * blockDim.y + threadIdx.y;
  if (row >= MROWS) return;
  float sc = rsqrtf(mv[2 * c + 1] + 1e-5f) * g[c];
  float sh = bb[c] - mv[2 * c] * sc;
  float mx = -3.4e38f;
  for (int k = 0; k < K_; ++k) {
    float val = fmaxf(e[(row * K_ + k) * C_ + c] * sc + sh, 0.f);
    mx = fmaxf(mx, val);
  }
  out[row * C_ + c] = mx;
}

// ---------------------------------------------------------------------------
// GAT attention + aggregation (one wave32 per point; lane == channel)
// ---------------------------------------------------------------------------
__device__ __forceinline__ float wsum32(float v) {
#pragma unroll
  for (int m = 16; m > 0; m >>= 1) v += __shfl_xor(v, m, 32);
  return v;
}

__global__ void gat_kernel(const float* __restrict__ h,
                           const int* __restrict__ idx,
                           const float* __restrict__ asrc,
                           const float* __restrict__ adst,
                           float* __restrict__ out)
{
  int c = threadIdx.x;
  long long row = (long long)blockIdx.x * blockDim.y + threadIdx.y;
  if (row >= MROWS) return;
  int b = (int)(row / N_);
  float hv = h[row * C_ + c];
  float ssrc = wsum32(hv * asrc[c]);
  float ek[K_], hk[K_];
  float mx = -3.4e38f;
  for (int k = 0; k < K_; ++k) {
    int nk = idx[row * K_ + k];
    float hn = h[((long long)b * N_ + nk) * C_ + c];
    hk[k] = hn;
    float sd = wsum32(hn * adst[c]);
    float er = ssrc + sd;
    float ev = er > 0.f ? er : 0.2f * er;     // leaky_relu(0.2)
    ek[k] = ev;
    mx = fmaxf(mx, ev);
  }
  float den = 0.f, acc = 0.f;
  for (int k = 0; k < K_; ++k) {
    float a = expf(ek[k] - mx);
    den += a;
    acc += a * hk[k];
  }
  out[row * C_ + c] = acc / den;
}

// gcn_in = (x + nbsum)/10 ; gin_in = (1+eps)*x + nbsum
__global__ void gcngin_build(const float* __restrict__ x,
                             const float* __restrict__ nbs,
                             const float* __restrict__ eps,
                             float* __restrict__ gcn_in,
                             float* __restrict__ gin_in)
{
  long long i = (long long)blockIdx.x * blockDim.x + threadIdx.x;
  if (i >= MROWS * C_) return;
  float xv = x[i], s = nbs[i];
  gcn_in[i] = (xv + s) * 0.1f;
  gin_in[i] = (1.f + eps[0]) * xv + s;
}

// snew (+)= w1*x + w2*o2 + ... + w7*o7  (w0 multiplies zeros)
__global__ void mix_accum(float* __restrict__ snew, const float* __restrict__ x,
                          const float* __restrict__ o2, const float* __restrict__ o3,
                          const float* __restrict__ o4, const float* __restrict__ o5,
                          const float* __restrict__ o6, const float* __restrict__ o7,
                          const float* __restrict__ w, int init)
{
  long long i = (long long)blockIdx.x * blockDim.x + threadIdx.x;
  if (i >= MROWS * C_) return;
  float acc = init ? 0.f : snew[i];
  acc += w[1] * x[i]  + w[2] * o2[i] + w[3] * o3[i] + w[4] * o4[i]
       + w[5] * o5[i] + w[6] * o6[i] + w[7] * o7[i];
  snew[i] = acc;
}

__global__ void alphas_softmax(const float* __restrict__ a, float* __restrict__ wm)
{
  int e = threadIdx.x;
  if (e >= 14) return;
  float mx = -3.4e38f;
  for (int i = 0; i < 8; ++i) mx = fmaxf(mx, a[e * 8 + i]);
  float t[8], s = 0.f;
  for (int i = 0; i < 8; ++i) { t[i] = expf(a[e * 8 + i] - mx); s += t[i]; }
  for (int i = 0; i < 8; ++i) wm[e * 8 + i] = t[i] / s;
}

__global__ void copy_ch(const float* __restrict__ src, int srcC,
                        float* __restrict__ dst, int dstC, int doff, long long M)
{
  long long i = (long long)blockIdx.x * blockDim.x + threadIdx.x;
  if (i >= M * srcC) return;
  int c = (int)(i % srcC);
  long long r = i / srcC;
  dst[r * dstC + doff + c] = src[i];
}

// per-batch max & mean pooling over N points -> h[b, 0:C]=max, h[b, C:2C]=mean
__global__ void pool_maxmean(const float* __restrict__ f, int C,
                             float* __restrict__ hout)
{
  int b = blockIdx.y;
  int c = blockIdx.x * blockDim.x + threadIdx.x;
  if (c >= C) return;
  float mx = -3.4e38f, s = 0.f;
  for (int n = 0; n < N_; ++n) {
    float v = f[((long long)b * N_ + n) * C + c];
    mx = fmaxf(mx, v);
    s += v;
  }
  hout[(long long)b * 2 * C + c]     = mx;
  hout[(long long)b * 2 * C + C + c] = s / (float)N_;
}

__global__ void add_bias(float* __restrict__ y, long long M, int C,
                         const float* __restrict__ bias)
{
  long long i = (long long)blockIdx.x * blockDim.x + threadIdx.x;
  if (i >= M * C) return;
  y[i] += bias[i % C];
}

// final: out[b,c] = yp[b*48+c] + bias[c]  (un-pad 48 -> 40 logits)
__global__ void out_bias_copy(const float* __restrict__ yp,
                              const float* __restrict__ bias,
                              float* __restrict__ out)
{
  int i = blockIdx.x * blockDim.x + threadIdx.x;
  if (i >= 16 * 40) return;
  int b = i / 40, c = i % 40;
  out[i] = yp[b * 48 + c] + bias[c];
}

// ---------------------------------------------------------------------------
// Host-side orchestration.
// Input leaf order assumption: jax-tree flatten (dict keys sorted, lists in
// order). Leaves: alphas | cells[0..1]{edges[0..13]{c1x1(W,b,g), edge(W,b,g),
// gat(W,a_dst,a_src,b,g), gcn(W,b,g), gin(W,b,eps,g), mr(W,b,g)}, pp0(W,b,g),
// pp1(W,b,g)} | cls{c1(W,b,bias,g), c2(W,b,bias,g), c3(W,bias)} |
// fusion(W,b,g) | pre(W,b,g) | stem(W,b,g) | x   -> 621 leaves.
// ---------------------------------------------------------------------------
extern "C" void kernel_launch(void* const* d_in, const int* in_sizes, int n_in,
                              void* d_out, int out_size, void* d_ws, size_t ws_size,
                              hipStream_t stream)
{
  (void)in_sizes; (void)n_in; (void)out_size; (void)ws_size;
  auto F = [&](int i) -> const float* { return (const float*)d_in[i]; };
  const long long M = MROWS;

  // -------- workspace carving
  char* wsb = (char*)d_ws;
  size_t off = 0;
  auto af = [&](size_t n) -> float* {           // n fp32/dword elements
    float* p = (float*)(wsb + off);
    off = (off + n * sizeof(float) + 255) & ~(size_t)255;
    return p;
  };
  float* Wm    = af(14 * 8);
  float* mv    = af(2 * 1024);
  float* sq    = af(M);
  float* sbuf  = af(M * 96);
  int*   idxA  = (int*)af(M * K_);
  int*   idxB  = (int*)af(M * K_);
  float* st[6];
  for (int i = 0; i < 6; ++i) st[i] = af(M * C_);
  float* pre1  = af(M * 128);
  float* pre2  = af(M * 128);
  float* nbsum = af(M * C_);
  float* mdiff = af(M * C_);
  float* t1    = af(M * C_);
  float* tet   = af(M * C_);
  float* teb   = af(M * C_);
  float* tmr   = af(M * C_);
  float* hg    = af(M * C_);
  float* og    = af(M * C_);
  float* gcni  = af(M * C_);
  float* gini  = af(M * C_);
  float* ogcn  = af(M * C_);
  float* ogin  = af(M * C_);
  float* oedg  = af(M * C_);
  float* ebuf  = af(M * K_ * C_);
  float* fin   = af(M * 352);
  float* fout  = af(M * 1024);
  float* hp    = af(16 * 2048);
  float* h1    = af(16 * 512);
  float* h2    = af(16 * 256);
  float* h3p   = af(16 * 48);                       // padded logits (N=48)
  unsigned int* apA = (unsigned int*)af(M * 176);   // packed A, up to K=352
  unsigned int* apT = (unsigned int*)af(M * 16);    // packed A, K=32 temporaries
  unsigned int* wp  = (unsigned int*)af(524288);    // packed W, up to 2048x512

  // -------- helpers
  auto packA = [&](const float* A, int Ksrc, int Kp, long long Mm,
                   unsigned int* dst) {
    long long tot = Mm * (long long)(Kp >> 5) * 16;
    pack_a<<<(unsigned)((tot + 255) / 256), 256, 0, stream>>>(A, Ksrc, Kp, Mm, dst);
  };
  auto packW = [&](const float* W, int Ksrc, int Kp, int Nsrc, int Np) {
    long long tot = (long long)(Kp >> 5) * Np * 16;
    pack_w<<<(unsigned)((tot + 255) / 256), 256, 0, stream>>>(W, Ksrc, Kp, Nsrc, Np, wp);
  };
  auto gemmP = [&](const unsigned int* Ap, float* Y, long long Mm, int K, int N,
                   int acc) {
    dim3 grid(N / 16, (unsigned)((Mm + 63) / 64));
    gemm_wmma_packed<<<grid, dim3(32, 4), 0, stream>>>(Ap, wp, Y, (int)Mm, K, N, acc);
  };
  auto bnorm = [&](float* Y, long long Mm, int C, const float* g,
                   const float* b, int act) {
    bn_stats<<<C, 256, 0, stream>>>(Y, Mm, C, mv);
    long long tot = Mm * (long long)C;
    bn_apply<<<(unsigned)((tot + 255) / 256), 256, 0, stream>>>(Y, Mm, C, mv, g, b, act);
  };

  // -------- input leaf indices
  const int I_ALPHAS = 0;
  auto eBase  = [&](int ci, int e) { return 1 + ci * 300 + e * 21; };
  auto ppBase = [&](int ci) { return 1 + ci * 300 + 294; };
  const int I_CLS  = 601;   // c1: W,b,bias,g | c2: W,b,bias,g | c3: W,bias
  const int I_FUS  = 611;   // W,b,g
  const int I_STEM = 617;   // W,b,g
  const int I_X    = 620;

  alphas_softmax<<<1, 32, 0, stream>>>(F(I_ALPHAS), Wm);

  // -------- stem: (B,N,3) @ (3,96) -> BN (no act); K padded 3 -> 32
  packA(F(I_X), 3, 32, M, apA);
  packW(F(I_STEM + 0), 3, 32, 96, 96);
  gemmP(apA, sbuf, M, 32, 96, 0);
  bnorm(sbuf, M, 96, F(I_STEM + 2), F(I_STEM + 1), 0);

  // -------- kNN on stem features (both cells key off the same s0)
  rowsq_kernel<<<(unsigned)((M + 255) / 256), 256, 0, stream>>>(sbuf, 96, sq);
  knn_kernel<<<dim3(N_, B_), 128, 0, stream>>>(sbuf, sq, 96, 9, 1, idxA);
  knn_kernel<<<dim3(N_, B_), 128, 0, stream>>>(sbuf, sq, 96, 18, 2, idxB);

  const unsigned rows8 = (unsigned)((M + 7) / 8);
  const unsigned blk32 = (unsigned)((M * C_ + 255) / 256);

  auto run_mixed = [&](const float* x, const int* idx, int eb, int eidx,
                       float* snew, int init) {
    nb_agg<<<rows8, dim3(32, 8), 0, stream>>>(x, idx, nbsum, mdiff);
    packA(x, 32, 32, M, apA);                 // shared by 5 GEMMs below
    // c1x1
    packW(F(eb + 0), 32, 32, 32, 32);
    gemmP(apA, t1, M, 32, 32, 0);
    bnorm(t1, M, 32, F(eb + 2), F(eb + 1), 1);
    // edge conv (split-W gather identity)
    packW(F(eb + 3), 32, 32, 32, 32);
    gemmP(apA, tet, M, 32, 32, 0);
    packW(F(eb + 3) + 32 * 32, 32, 32, 32, 32);
    gemmP(apA, teb, M, 32, 32, 0);
    long long etot = M * K_ * C_;
    edge_build<<<(unsigned)((etot + 255) / 256), 256, 0, stream>>>(tet, teb, idx, ebuf);
    bn_stats<<<32, 256, 0, stream>>>(ebuf, M * K_, 32, mv);
    edge_bn_max<<<rows8, dim3(32, 8), 0, stream>>>(ebuf, mv, F(eb + 5), F(eb + 4), oedg);
    // mr: x@Wtop + maxdiff@Wbot
    packW(F(eb + 18), 32, 32, 32, 32);
    gemmP(apA, tmr, M, 32, 32, 0);
    packA(mdiff, 32, 32, M, apT);
    packW(F(eb + 18) + 32 * 32, 32, 32, 32, 32);
    gemmP(apT, tmr, M, 32, 32, 1);
    bnorm(tmr, M, 32, F(eb + 20), F(eb + 19), 1);
    // gat
    packW(F(eb + 6), 32, 32, 32, 32);
    gemmP(apA, hg, M, 32, 32, 0);
    gat_kernel<<<rows8, dim3(32, 8), 0, stream>>>(hg, idx, F(eb + 8), F(eb + 7), og);
    bnorm(og, M, 32, F(eb + 10), F(eb + 9), 1);
    // gcn / gin
    gcngin_build<<<blk32, 256, 0, stream>>>(x, nbsum, F(eb + 16), gcni, gini);
    packA(gcni, 32, 32, M, apT);
    packW(F(eb + 11), 32, 32, 32, 32);
    gemmP(apT, ogcn, M, 32, 32, 0);
    bnorm(ogcn, M, 32, F(eb + 13), F(eb + 12), 1);
    packA(gini, 32, 32, M, apT);
    packW(F(eb + 14), 32, 32, 32, 32);
    gemmP(apT, ogin, M, 32, 32, 0);
    bnorm(ogin, M, 32, F(eb + 17), F(eb + 15), 1);
    // weighted accumulate into snew
    mix_accum<<<blk32, 256, 0, stream>>>(snew, x, t1, oedg, tmr, og, ogcn, ogin,
                                         Wm + eidx * 8, init);
  };

  // -------- cells
  const float* s0 = sbuf; int s0C = 96;
  const float* s1 = sbuf; int s1C = 96;
  for (int ci = 0; ci < 2; ++ci) {
    const int* idx = (ci == 0) ? idxA : idxB;
    int pb = ppBase(ci);
    packA(s0, s0C, s0C, M, apA);
    packW(F(pb + 0), s0C, s0C, 32, 32);
    gemmP(apA, st[0], M, s0C, 32, 0);
    bnorm(st[0], M, 32, F(pb + 2), F(pb + 1), 1);
    packA(s1, s1C, s1C, M, apA);
    packW(F(pb + 3), s1C, s1C, 32, 32);
    gemmP(apA, st[1], M, s1C, 32, 0);
    bnorm(st[1], M, 32, F(pb + 5), F(pb + 4), 1);

    int offE = 0;
    for (int step = 0; step < 4; ++step) {
      int ns = 2 + step;
      float* snew = st[2 + step];
      for (int j = 0; j < ns; ++j) {
        int eidx = offE + j;
        run_mixed(st[j], idx, eBase(ci, eidx), eidx, snew, j == 0);
      }
      offE += ns;
    }
    float* preb = (ci == 0) ? pre1 : pre2;
    for (int t = 0; t < 4; ++t)
      copy_ch<<<blk32, 256, 0, stream>>>(st[2 + t], 32, preb, 128, 32 * t, M);
    s0 = s1; s0C = s1C;        // (s0, s1) = (old s1, concat)
    s1 = preb; s1C = 128;
  }

  // -------- fusion: concat(96+128+128)=352 -> 1024, BN, leaky
  copy_ch<<<(unsigned)((M * 96 + 255) / 256), 256, 0, stream>>>(sbuf, 96, fin, 352, 0, M);
  copy_ch<<<(unsigned)((M * 128 + 255) / 256), 256, 0, stream>>>(pre1, 128, fin, 352, 96, M);
  copy_ch<<<(unsigned)((M * 128 + 255) / 256), 256, 0, stream>>>(pre2, 128, fin, 352, 224, M);
  packA(fin, 352, 352, M, apA);
  packW(F(I_FUS + 0), 352, 352, 1024, 1024);
  gemmP(apA, fout, M, 352, 1024, 0);
  bnorm(fout, M, 1024, F(I_FUS + 2), F(I_FUS + 1), 2);
  pool_maxmean<<<dim3(4, B_), 256, 0, stream>>>(fout, 1024, hp);

  // -------- classifier head
  packA(hp, 2048, 2048, 16, apA);
  packW(F(I_CLS + 0), 2048, 2048, 512, 512);
  gemmP(apA, h1, 16, 2048, 512, 0);
  add_bias<<<(16 * 512 + 255) / 256, 256, 0, stream>>>(h1, 16, 512, F(I_CLS + 2));
  bnorm(h1, 16, 512, F(I_CLS + 3), F(I_CLS + 1), 2);

  packA(h1, 512, 512, 16, apA);
  packW(F(I_CLS + 4), 512, 512, 256, 256);
  gemmP(apA, h2, 16, 512, 256, 0);
  add_bias<<<(16 * 256 + 255) / 256, 256, 0, stream>>>(h2, 16, 256, F(I_CLS + 6));
  bnorm(h2, 16, 256, F(I_CLS + 7), F(I_CLS + 5), 2);

  // final 256 -> 40 (N padded to 48), then un-pad + bias into d_out
  packA(h2, 256, 256, 16, apA);
  packW(F(I_CLS + 8), 256, 256, 40, 48);
  gemmP(apA, h3p, 16, 256, 48, 0);
  out_bias_copy<<<(16 * 40 + 255) / 256, 256, 0, stream>>>(h3p, F(I_CLS + 9),
                                                           (float*)d_out);
}
